// Synthesizer_63075889709470
// MI455X (gfx1250) — matrix operand order repaired and checked by reading, not compiled, not run on hardware
//
#include <hip/hip_runtime.h>
#include <stdint.h>

#define PACKED_STRIDE 416
#define N_SAMPLES     32768
#define FFT_N         16384
#define PI_F          3.14159265358979f
#define NYQF          11025.0f
#define MIN_F0_C      (40.0f / 11025.0f)
#define F0_SPAN_C     (3000.0f / 11025.0f - 40.0f / 11025.0f)

typedef __attribute__((ext_vector_type(16))) _Float16 v16h;
typedef __attribute__((ext_vector_type(8)))  _Float16 v8h;
typedef __attribute__((ext_vector_type(8)))  float    v8f;

// ---------------------------------------------------------------------------
// Kernel A0: softmax over packed[:,0:128] for all 128 events, then the
// 128x128 @ 128 GEMV against `frequencies` via v_wmma_f32_16x16x32_f16.
// Single wave (wave32) => EXEC is all-ones around every WMMA.
// Fragment fills are branchless: A uses k = 16h + s, one contiguous
// 16-half run per lane => one 32B LDS load per A fragment; the
// B fragment (frequencies in column N=0) is one contiguous 32B LDS load
// scaled by a per-lane 0/1 f16 factor.
// ---------------------------------------------------------------------------
__global__ __launch_bounds__(32) void k_f0_softmax_wmma(
    const float* __restrict__ packed, const float* __restrict__ freqs,
    float* __restrict__ ws_f0) {
  __shared__ _Float16 shW[128][128];   // softmax weights, f16 (32 KB)
  __shared__ _Float16 shF[128];        // frequencies, f16
  const int lane = threadIdx.x;

  for (int j = lane; j < 128; j += 32) shF[j] = (_Float16)freqs[j];

  // each lane owns 4 rows (events)
  for (int r = lane; r < 128; r += 32) {
    const float* row = packed + r * PACKED_STRIDE;
    float m = -3.4e38f;
    for (int j = 0; j < 128; ++j) m = fmaxf(m, row[j]);
    float s = 0.f;
    for (int j = 0; j < 128; ++j) {
      float e = __expf(row[j] - m);
      s += e;
      shW[r][j] = (_Float16)e;
    }
    float inv = 1.0f / s;
    for (int j = 0; j < 128; ++j)
      shW[r][j] = (_Float16)((float)shW[r][j] * inv);
  }
  __syncthreads();

  const int lm    = lane & 15;
  const int koffA = (lane & 16);       // 0 (lanes 0-15) or 16 halves (lanes 16-31)
  const int koffB = (lane & 16);       // 0 or 16 halves
  const _Float16 selh = (lm == 0) ? (_Float16)1.0f : (_Float16)0.0f;

  #pragma unroll
  for (int m = 0; m < 8; ++m) {        // 16-row tile of events
    v8f acc = {0.f, 0.f, 0.f, 0.f, 0.f, 0.f, 0.f, 0.f};
    const _Float16* rowp = &shW[m * 16 + lm][0];
    #pragma unroll
    for (int kt = 0; kt < 4; ++kt) {   // K tiles of 32
      const _Float16* pA = rowp + kt * 32 + koffA;
      v16h a = *(const v16h*)(pA);         // K = kt*32 + koffA + [0,16)
      v16h ball = *(const v16h*)(&shF[kt * 32 + koffB]);
      v16h bf = ball * selh;               // column N=0 only (lanes with lm==0)
      acc = __builtin_amdgcn_wmma_f32_16x16x32_f16(false, a, false, bf,
                                                   (short)0, acc, false, false);
    }
    // D column 0: lane 0 holds rows m*16+0..7 in acc[0..7], lane 16 rows m*16+8..15
    if (lane == 0) {
      #pragma unroll
      for (int i = 0; i < 8; ++i) ws_f0[m * 16 + i] = acc[i];
    }
    if (lane == 16) {
      #pragma unroll
      for (int i = 0; i < 8; ++i) ws_f0[m * 16 + 8 + i] = acc[i];
    }
  }
}

// ---------------------------------------------------------------------------
// Kernel A1: per-event phase steps r[b,h] and the 128-frame amplitude scan.
// f0 is constant across frames, so interp(radians) is a constant => pure
// phase ramp; scan is per-harmonic independent (lane h runs 128 steps).
// ---------------------------------------------------------------------------
__global__ __launch_bounds__(32) void k_prep(
    const float* __restrict__ packed, const float* __restrict__ ws_f0,
    float* __restrict__ ws_r, float* __restrict__ ws_res) {
  const int b = blockIdx.x;
  const int h = threadIdx.x;
  if (h >= 8) return;
  const float* row = packed + b * PACKED_STRIDE;
  float f0s = ws_f0[b];
  float f0f = MIN_F0_C + (f0s * f0s * NYQF) * F0_SPAN_C;
  float r = f0f * (float)(h + 1) * (PI_F / NYQF);
  if (r >= PI_F) r = 0.f;
  ws_r[b * 8 + h] = r;

  float hamp = row[384 + h];               // SL_HAMP
  float hdec = 0.5f + row[392 + h] * 0.5f; // SL_HDEC: MIN_RES + p*RES_SPAN
  float cur = 0.f;
  float* res = ws_res + (b * 8 + h) * 128;
  for (int f = 0; f < 128; ++f) {
    float a = row[256 + f] * 2.f - 1.f;    // SL_AMP
    float c = fmaxf(cur + a * hamp, 0.f);
    res[f] = c;
    cur = c * hdec;
  }
}

// ---------------------------------------------------------------------------
// Kernel B: per-event filtered noise via in-LDS FFT.
//   - async global->LDS b64 loads with bit-reversal folded into the per-lane
//     LDS destination address (ASYNCcnt path)
//   - 16384-pt complex FFT (packed real 32768), untangle * H * repack,
//     inverse DIF FFT, scale 1/N, multiply amp envelope, store to ws.
// LDS: 16384 float2 = 128 KB (fits CDNA5's 320KB/WGP).
// ---------------------------------------------------------------------------
__device__ __forceinline__ float h_filt(const float* shNF, int k) {
  float c = ((float)k + 0.5f) * (1.0f / 1024.0f) - 0.5f;  // interp 16 -> 16384
  c = fminf(fmaxf(c, 0.f), 15.f);
  int   i0 = (int)c;
  int   i1 = (i0 < 15) ? i0 + 1 : 15;
  float w  = c - (float)i0;
  return shNF[i0] + (shNF[i1] - shNF[i0]) * w;
}
__device__ __forceinline__ float amp_env(const float* shAmp, int t) {
  float c = ((float)t + 0.5f) * (1.0f / 256.0f) - 0.5f;   // interp 128 -> 32768
  c = fminf(fmaxf(c, 0.f), 127.f);
  int   i0 = (int)c;
  int   i1 = (i0 < 127) ? i0 + 1 : 127;
  float w  = c - (float)i0;
  float v  = shAmp[i0] + (shAmp[i1] - shAmp[i0]) * w;
  return v >= 0.f ? v : 0.2f * v;
}

__global__ __launch_bounds__(256) void k_noise_fft(
    const float* __restrict__ packed, const float* __restrict__ noise,
    const float* __restrict__ global_amp, float* __restrict__ ws_noise) {
  __shared__ float2 Z[FFT_N];      // 128 KB
  __shared__ float  shNF[16];
  __shared__ float  shAmp[128];
  const int tid = threadIdx.x;
  const int b   = blockIdx.x;

  const float ga = global_amp[0];
  if (tid < 16)  shNF[tid]  = packed[b * PACKED_STRIDE + 400 + tid] * ga; // fold global_amp into H
  if (tid < 128) shAmp[tid] = packed[b * PACKED_STRIDE + 256 + tid] * 2.f - 1.f;

  // ---- async global->LDS load, bit-reversed complex packing z[n]=x[2n]+i x[2n+1]
  const float* src = noise + (size_t)b * N_SAMPLES;
  uint32_t zbase = (uint32_t)(unsigned long long)&Z[0];
  for (int i = 0; i < 64; ++i) {
    uint32_t n    = (uint32_t)(i * 256 + tid);
    uint32_t voff = n * 8u;                                 // byte offset of x[2n]
    uint32_t lds  = zbase + ((__brev(n) >> 18) * 8u);       // bitrev14(n)
    asm volatile("global_load_async_to_lds_b64 %0, %1, %2"
                 :: "v"(lds), "v"(voff), "s"(src) : "memory");
  }
  asm volatile("s_wait_asynccnt 0x0" ::: "memory");
  __syncthreads();

  // ---- forward FFT: radix-2 DIT (bit-reversed in -> natural out), sign -1
  for (int s = 1; s <= 14; ++s) {
    const int half = 1 << (s - 1);
    const float astep = -PI_F / (float)half;
    for (int t = tid; t < FFT_N / 2; t += 256) {
      int j  = t & (half - 1);
      int i0 = ((t >> (s - 1)) << s) + j;
      int i1 = i0 + half;
      float sn, cs;
      __sincosf(astep * (float)j, &sn, &cs);
      float2 a = Z[i0], bb = Z[i1];
      float2 bw = make_float2(bb.x * cs - bb.y * sn, bb.x * sn + bb.y * cs);
      Z[i0] = make_float2(a.x + bw.x, a.y + bw.y);
      Z[i1] = make_float2(a.x - bw.x, a.y - bw.y);
    }
    __syncthreads();
  }

  // ---- untangle rfft bins, multiply real filter H, repack for real inverse
  for (int k = tid; k <= FFT_N / 2; k += 256) {
    if (k == 0) {
      float2 z0 = Z[0];
      float X0 = z0.x + z0.y;          // DC bin
      float Y0 = h_filt(shNF, 0) * X0; // Y[N] = H[16384]*XN = 0 (padded zero)
      Z[0] = make_float2(0.5f * Y0, 0.5f * Y0);
    } else if (k == FFT_N / 2) {
      float Hm = h_filt(shNF, k);
      float2 zm = Z[k];                // Y'[N/2] = conj(H*conj(Z)) = H*Z
      Z[k] = make_float2(Hm * zm.x, Hm * zm.y);
    } else {
      float2 Za = Z[k], Zb = Z[FFT_N - k];
      float hrx = 0.5f * (Za.x + Zb.x), hry = 0.5f * (Za.y - Zb.y);
      float hdx = 0.5f * (Za.x - Zb.x), hdy = 0.5f * (Za.y + Zb.y);
      float sw, cw;
      __sincosf(PI_F * (float)k * (1.0f / (float)FFT_N), &sw, &cw);
      float wx = cw * hdx + sw * hdy, wy = cw * hdy - sw * hdx;
      float Xkx = hrx + wy, Xky =  hry - wx;     // X[k]
      float Xnx = hrx - wy, Xny = -hry - wx;     // X[N-k]
      float Hk = h_filt(shNF, k), Hn = h_filt(shNF, FFT_N - k);
      float Ykx = Hk * Xkx, Yky = Hk * Xky;
      float Ynx = Hn * Xnx, Yny = Hn * Xny;
      float prx = 0.5f * (Ykx + Ynx), pry = 0.5f * (Yky - Yny);
      float pdx = 0.5f * (Ykx - Ynx), pdy = 0.5f * (Yky + Yny);
      float vx = cw * pdx - sw * pdy, vy = cw * pdy + sw * pdx;
      Z[k]         = make_float2(prx - vy,  pry + vx);
      Z[FFT_N - k] = make_float2(prx + vy, -pry + vx);
    }
  }
  __syncthreads();

  // ---- inverse FFT: radix-2 DIF (natural in -> bit-reversed out), sign +1
  for (int s = 14; s >= 1; --s) {
    const int half = 1 << (s - 1);
    const float astep = PI_F / (float)half;
    for (int t = tid; t < FFT_N / 2; t += 256) {
      int j  = t & (half - 1);
      int i0 = ((t >> (s - 1)) << s) + j;
      int i1 = i0 + half;
      float2 a = Z[i0], bb = Z[i1];
      Z[i0] = make_float2(a.x + bb.x, a.y + bb.y);
      float dx = a.x - bb.x, dy = a.y - bb.y;
      float sn, cs;
      __sincosf(astep * (float)j, &sn, &cs);
      Z[i1] = make_float2(dx * cs - dy * sn, dx * sn + dy * cs);
    }
    __syncthreads();
  }

  // ---- scale 1/N, apply amp envelope, store time samples (coalesced float2)
  const float invN = 1.0f / (float)FFT_N;
  float2* outp = (float2*)(ws_noise + (size_t)b * N_SAMPLES);
  for (int n = tid; n < FFT_N; n += 256) {
    int ir = (int)(__brev((uint32_t)n) >> 18);
    float2 v = Z[ir];
    float y0 = v.x * invN, y1 = v.y * invN;
    outp[n] = make_float2(y0 * amp_env(shAmp, 2 * n),
                          y1 * amp_env(shAmp, 2 * n + 1));
  }
}

// ---------------------------------------------------------------------------
// Kernel C: oscillator bank + final sum over 8 events x 8 harmonics.
// noise_out broadcast over 8 harmonics then /8 cancels under the sum.
// ---------------------------------------------------------------------------
__global__ __launch_bounds__(256) void k_mix(
    const float* __restrict__ ws_r, const float* __restrict__ ws_res,
    const float* __restrict__ ws_noise, const float* __restrict__ osc_amp,
    float* __restrict__ out) {
  const int t = blockIdx.x * 256 + threadIdx.x;
  const int g = blockIdx.y;
  const float oa = osc_amp[0];

  float c = ((float)t + 0.5f) * (1.0f / 256.0f) - 0.5f;  // res/amp frame coord
  c = fminf(fmaxf(c, 0.f), 127.f);
  int   i0 = (int)c;
  int   i1 = (i0 < 127) ? i0 + 1 : 127;
  float w  = c - (float)i0;
  float tp1 = (float)(t + 1);

  float sum = 0.f;
  for (int e = 0; e < 8; ++e) {
    int b = g * 8 + e;
    sum += ws_noise[(size_t)b * N_SAMPLES + t];
    const float* rr  = ws_r + b * 8;
    const float* res = ws_res + (size_t)b * 8 * 128;
    #pragma unroll
    for (int h = 0; h < 8; ++h) {
      float r0 = res[h * 128 + i0];
      float rv = r0 + (res[h * 128 + i1] - r0) * w;
      sum += rv * __sinf(rr[h] * tp1) * oa;
    }
  }
  out[(size_t)g * N_SAMPLES + t] = sum;
}

// ---------------------------------------------------------------------------
// Workspace layout (floats):
//   [0,128)          ws_f0
//   [128,1152)       ws_r    (128 events x 8 harmonics)
//   [1152,132224)    ws_res  (128 x 8 x 128 frames)
//   [132224, +4.19M) ws_noise(128 x 32768)   => ~17.3 MB total
// ---------------------------------------------------------------------------
extern "C" void kernel_launch(void* const* d_in, const int* in_sizes, int n_in,
                              void* d_out, int out_size, void* d_ws, size_t ws_size,
                              hipStream_t stream) {
  (void)in_sizes; (void)n_in; (void)out_size; (void)ws_size;
  const float* packed = (const float*)d_in[0];
  const float* freqs  = (const float*)d_in[1];
  const float* gamp   = (const float*)d_in[2];
  const float* oamp   = (const float*)d_in[3];
  const float* noise  = (const float*)d_in[4];
  float* out = (float*)d_out;

  float* ws       = (float*)d_ws;
  float* ws_f0    = ws;
  float* ws_r     = ws + 128;
  float* ws_res   = ws + 1152;
  float* ws_noise = ws + 132224;

  k_f0_softmax_wmma<<<1, 32, 0, stream>>>(packed, freqs, ws_f0);
  k_prep<<<128, 32, 0, stream>>>(packed, ws_f0, ws_r, ws_res);
  k_noise_fft<<<128, 256, 0, stream>>>(packed, noise, gamp, ws_noise);
  k_mix<<<dim3(N_SAMPLES / 256, 16), 256, 0, stream>>>(ws_r, ws_res, ws_noise, oamp, out);
}
